// GAT_62689342652836
// MI455X (gfx1250) — matrix operand order repaired and checked by reading, not compiled, not run on hardware
//
#include <hip/hip_runtime.h>
#include <hip/hip_bf16.h>
#include <math.h>

// ---------------------------------------------------------------------------
// 2-layer GAT for MI455X (gfx1250, wave32).
// GEMMs: bf16 WMMA (v_wmma_f32_16x16x32_bf16), f32 accumulate.
//  - A fragments: direct global->register (4x b128 + v_cvt_pk_bf16_f32),
//    software-pipelined one K-step ahead so WMMAs hide the load latency.
//  - B slab: staged once to LDS column-major+padded so each lane fragment is
//    one contiguous 32B run (2x ds_load_b128); 4 fragments issued ahead of the
//    WMMA quartet so dscnt waits are partial. Single barrier in the kernel.
// Edge phase: f32, wave-per-edge float4 gather + global f32 atomics
// (h is 51MB < 192MB L2, so gather+scatter stay L2-resident).
// ---------------------------------------------------------------------------

typedef __attribute__((ext_vector_type(16))) __bf16 v16bf;
typedef __attribute__((ext_vector_type(2)))  __bf16 v2bf;
typedef __attribute__((ext_vector_type(8)))  float  v8f;

#define FIN        256     // inner dim for both layers (Fin and H*Hd)
#define NEG_SLOPE  0.2f
#define BN_EPS     1e-5f

#define B_STRIDE   272     // bf16 per B column (256 data + pad), 544B: 32B-aligned

__device__ __forceinline__ float leaky(float x) { return x > 0.f ? x : NEG_SLOPE * x; }

__device__ __forceinline__ void atomicMaxF(float* addr, float val) {
    int* ai = (int*)addr;
    int old = __float_as_int(*addr);
    while (__int_as_float(old) < val) {
        int assumed = old;
        old = atomicCAS(ai, assumed, __float_as_int(val));
        if (old == assumed) break;
    }
}

// ------------------------------ utility kernels ----------------------------

__global__ void k_fill(float* __restrict__ p, float v, int n) {
    int i = blockIdx.x * blockDim.x + threadIdx.x;
    if (i < n) p[i] = v;
}

__global__ void k_cvt_bf16(const float* __restrict__ src, __bf16* __restrict__ dst, int n) {
    int i = blockIdx.x * blockDim.x + threadIdx.x;
    if (i < n) dst[i] = (__bf16)src[i];
}

// ---------------------- fused linear + attention dots ----------------------
// H[N,Fout] = X[N,256] @ Wb[256,Fout]  (bf16 WMMA, f32 acc)
// a_s[n,h] += sum_c H[n,h,c]*att_s[h,c] over this block's 64 columns (atomic)
// Block: 256 threads = 8 waves. Tile: 128 rows x 64 cols. grid.y = Fout/64.
__global__ __launch_bounds__(256) void k_gemm_att(
    const float* __restrict__ X, const __bf16* __restrict__ Wb,
    int N, int Fout, int Chead,
    const float* __restrict__ att_s, const float* __restrict__ att_d,
    float* __restrict__ Hout, float* __restrict__ a_s, float* __restrict__ a_d)
{
    // B: 64 cols x 256 K, column-major (K contiguous per column), padded stride.
    __shared__ __attribute__((aligned(32))) __bf16 Bsw[64 * B_STRIDE];   // 34 KB

    const int tid  = threadIdx.x;
    const int lane = tid & 31, wave = tid >> 5;
    const int hi = lane >> 4, l16 = lane & 15;
    const int rowBase = blockIdx.x * 128;
    const int n0 = blockIdx.y * 64;

    // Stage weight slab column-major: Bsw[col*B_STRIDE + k] = Wb[k*Fout + n0+col]
    // Packed v2bf stores along K; global reads coalesced across cols.
    for (int e = tid; e < 64 * 128; e += 256) {
        int col = e & 63;
        int kp  = (e >> 6) * 2;
        v2bf p;
        p.x = (__bf16)Wb[(size_t)kp * Fout + n0 + col];
        p.y = (__bf16)Wb[(size_t)(kp + 1) * Fout + n0 + col];
        *(v2bf*)&Bsw[col * B_STRIDE + kp] = p;
    }
    __syncthreads();   // only barrier in the kernel

    // Each lane owns one A row: (l16 selects row, hi selects K-chunks).
    // Clamp OOB rows to a valid address; their C rows are guarded at store.
    const int myRow = rowBase + wave * 16 + l16;
    const int gr = myRow < N ? myRow : N - 1;
    const float* __restrict__ xrow = X + (size_t)gr * FIN + hi * 16;
    const __bf16* __restrict__ bcol = &Bsw[l16 * B_STRIDE + hi * 16];

    v8f acc[4] = {};

    // prefetch K-step 0 of A (one 64B run: K = hi*16..+15)
    float4 p0 = *(const float4*)&xrow[0];
    float4 p1 = *(const float4*)&xrow[4];
    float4 p2 = *(const float4*)&xrow[8];
    float4 p3 = *(const float4*)&xrow[12];

    #pragma unroll
    for (int i = 0; i < FIN / 32; ++i) {
        const int kk = i * 32;
        // convert the prefetched A chunks to this step's fragment
        v16bf af;
        af[0]  = (__bf16)p0.x; af[1]  = (__bf16)p0.y; af[2]  = (__bf16)p0.z; af[3]  = (__bf16)p0.w;
        af[4]  = (__bf16)p1.x; af[5]  = (__bf16)p1.y; af[6]  = (__bf16)p1.z; af[7]  = (__bf16)p1.w;
        af[8]  = (__bf16)p2.x; af[9]  = (__bf16)p2.y; af[10] = (__bf16)p2.z; af[11] = (__bf16)p2.w;
        af[12] = (__bf16)p3.x; af[13] = (__bf16)p3.y; af[14] = (__bf16)p3.z; af[15] = (__bf16)p3.w;

        // issue next K-step's A loads; the 4 WMMAs below hide their latency
        if (i < FIN / 32 - 1) {
            p0 = *(const float4*)&xrow[kk + 32];
            p1 = *(const float4*)&xrow[kk + 36];
            p2 = *(const float4*)&xrow[kk + 40];
            p3 = *(const float4*)&xrow[kk + 44];
        }

        // all 4 B fragments in flight before the WMMA quartet (partial dscnt waits)
        v16bf b0 = *(const v16bf*)&bcol[kk];
        v16bf b1 = *(const v16bf*)&bcol[16 * B_STRIDE + kk];
        v16bf b2 = *(const v16bf*)&bcol[32 * B_STRIDE + kk];
        v16bf b3 = *(const v16bf*)&bcol[48 * B_STRIDE + kk];

        acc[0] = __builtin_amdgcn_wmma_f32_16x16x32_bf16(false, af, false, b0, (short)0, acc[0], false, false);
        acc[1] = __builtin_amdgcn_wmma_f32_16x16x32_bf16(false, af, false, b1, (short)0, acc[1], false, false);
        acc[2] = __builtin_amdgcn_wmma_f32_16x16x32_bf16(false, af, false, b2, (short)0, acc[2], false, false);
        acc[3] = __builtin_amdgcn_wmma_f32_16x16x32_bf16(false, af, false, b3, (short)0, acc[3], false, false);
    }

    // Epilogue: store H and fused attention dot products.
    // This 64-col block lies entirely inside one head (64 | Chead).
    const int head = n0 / Chead;
    const int cinh0 = (n0 % Chead) + l16;
    float asv[4], adv[4];
    #pragma unroll
    for (int nt = 0; nt < 4; ++nt) {
        asv[nt] = att_s[head * Chead + cinh0 + nt * 16];
        adv[nt] = att_d[head * Chead + cinh0 + nt * 16];
    }
    #pragma unroll
    for (int r = 0; r < 8; ++r) {
        // C/D layout: vgpr r -> row r (lanes 0-15) / row r+8 (lanes 16-31), col=l16
        int grow = rowBase + wave * 16 + hi * 8 + r;
        float ss = 0.f, sd = 0.f;
        if (grow < N) {
            #pragma unroll
            for (int nt = 0; nt < 4; ++nt) {
                float v = acc[nt][r];
                Hout[(size_t)grow * Fout + n0 + nt * 16 + l16] = v;
                ss += v * asv[nt];
                sd += v * adv[nt];
            }
        }
        // reduce across the 16 lanes of each half-wave (uniform within half)
        #pragma unroll
        for (int m = 8; m >= 1; m >>= 1) {
            ss += __shfl_xor(ss, m, 32);
            sd += __shfl_xor(sd, m, 32);
        }
        if (l16 == 0 && grow < N) {
            atomicAdd(&a_s[grow * 2 + head], ss);
            atomicAdd(&a_d[grow * 2 + head], sd);
        }
    }
}

// -------------------------- edge-phase kernels -----------------------------
// Self loops are synthesized on the fly: edge e >= E has src = dst = e - E.

__global__ void k_logit_max(const int* __restrict__ ei, int E, int N,
                            const float* __restrict__ a_s, const float* __restrict__ a_d,
                            float* __restrict__ m)
{
    int e = blockIdx.x * blockDim.x + threadIdx.x;
    int Etot = E + N;
    if (e >= Etot) return;
    int s = (e < E) ? ei[e]     : (e - E);
    int d = (e < E) ? ei[E + e] : (e - E);
    #pragma unroll
    for (int h = 0; h < 2; ++h) {
        float v = leaky(a_s[s * 2 + h] + a_d[d * 2 + h]);
        atomicMaxF(&m[d * 2 + h], v);
    }
}

__global__ void k_expsum(const int* __restrict__ ei, int E, int N,
                         const float* __restrict__ a_s, const float* __restrict__ a_d,
                         const float* __restrict__ m, float* __restrict__ ex,
                         float* __restrict__ ssum)
{
    int e = blockIdx.x * blockDim.x + threadIdx.x;
    int Etot = E + N;
    if (e >= Etot) return;
    int s = (e < E) ? ei[e]     : (e - E);
    int d = (e < E) ? ei[E + e] : (e - E);
    #pragma unroll
    for (int h = 0; h < 2; ++h) {
        float lg = leaky(a_s[s * 2 + h] + a_d[d * 2 + h]);
        float mm = m[d * 2 + h];
        if (!(mm > -1e30f)) mm = 0.f;               // non-finite guard (ref parity)
        float v = expf(lg - mm);
        ex[(size_t)e * 2 + h] = v;
        atomicAdd(&ssum[d * 2 + h], v);
    }
}

// One wave per edge; float4 gather of h[src], per-head alpha, atomic scatter.
__global__ __launch_bounds__(256) void k_aggregate(
    const int* __restrict__ ei, int E, int N, int Fout, int Chead,
    const float* __restrict__ Hfeat, const float* __restrict__ ex,
    const float* __restrict__ ssum, float* __restrict__ agg)
{
    int wave = blockIdx.x * (blockDim.x >> 5) + (threadIdx.x >> 5);
    int lane = threadIdx.x & 31;
    int Etot = E + N;
    if (wave >= Etot) return;
    int e = wave;
    int s = (e < E) ? ei[e]     : (e - E);
    int d = (e < E) ? ei[E + e] : (e - E);
    int nv4 = Fout >> 2;
    const float4* hsrc = (const float4*)(Hfeat + (size_t)s * Fout);
    float* dbase = agg + (size_t)d * Fout;
    for (int i = lane; i < nv4; i += 32) {
        int head = (i * 4) / Chead;
        float w = ex[(size_t)e * 2 + head] / (ssum[d * 2 + head] + 1e-16f);
        float4 v = hsrc[i];
        float* dp = dbase + i * 4;
        atomicAdd(dp + 0, w * v.x);
        atomicAdd(dp + 1, w * v.y);
        atomicAdd(dp + 2, w * v.z);
        atomicAdd(dp + 3, w * v.w);
    }
}

// layer-1 epilogue: +bias, BatchNorm(eval), ELU — in place
__global__ void k_post1(float* __restrict__ hb, const float* __restrict__ b,
                        const float* __restrict__ g, const float* __restrict__ be,
                        const float* __restrict__ mu, const float* __restrict__ var, int total)
{
    int i = blockIdx.x * blockDim.x + threadIdx.x;
    if (i >= total) return;
    int j = i & 255;
    float v = hb[i] + b[j];
    v = (v - mu[j]) * rsqrtf(var[j] + BN_EPS) * g[j] + be[j];
    hb[i] = v > 0.f ? v : expm1f(v);
}

// final: head mean + bias + log_softmax over 64 classes, one wave per node
__global__ __launch_bounds__(256) void k_final(const float* __restrict__ agg2,
                                               const float* __restrict__ b2,
                                               float* __restrict__ out, int N)
{
    int node = blockIdx.x * 8 + (threadIdx.x >> 5);
    int lane = threadIdx.x & 31;
    if (node >= N) return;
    const float* row = agg2 + (size_t)node * 128;   // [2][64]
    int c0 = lane, c1 = lane + 32;
    float v0 = 0.5f * (row[c0] + row[64 + c0]) + b2[c0];
    float v1 = 0.5f * (row[c1] + row[64 + c1]) + b2[c1];
    float mx = fmaxf(v0, v1);
    #pragma unroll
    for (int m = 16; m; m >>= 1) mx = fmaxf(mx, __shfl_xor(mx, m, 32));
    float se = expf(v0 - mx) + expf(v1 - mx);
    #pragma unroll
    for (int m = 16; m; m >>= 1) se += __shfl_xor(se, m, 32);
    float lse = mx + logf(se);
    out[(size_t)node * 64 + c0] = v0 - lse;
    out[(size_t)node * 64 + c1] = v1 - lse;
}

// ------------------------------- launcher ----------------------------------

extern "C" void kernel_launch(void* const* d_in, const int* in_sizes, int n_in,
                              void* d_out, int out_size, void* d_ws, size_t ws_size,
                              hipStream_t stream)
{
    const float* x   = (const float*)d_in[0];
    const int*   ei  = (const int*)d_in[1];
    const float* W1  = (const float*)d_in[2];
    const float* as1 = (const float*)d_in[3];
    const float* ad1 = (const float*)d_in[4];
    const float* b1  = (const float*)d_in[5];
    const float* bng = (const float*)d_in[6];
    const float* bnb = (const float*)d_in[7];
    const float* bnm = (const float*)d_in[8];
    const float* bnv = (const float*)d_in[9];
    const float* W2  = (const float*)d_in[10];
    const float* as2 = (const float*)d_in[11];
    const float* ad2 = (const float*)d_in[12];
    const float* b2  = (const float*)d_in[13];
    float* out = (float*)d_out;

    const int N    = in_sizes[0] / 256;
    const int E    = in_sizes[1] / 2;
    const int Etot = E + N;

    // workspace layout
    char* ws = (char*)d_ws;
    size_t off = 0;
    auto walloc = [&](size_t bytes) -> void* {
        void* p = ws + off;
        off += (bytes + 255) & ~(size_t)255;
        return p;
    };
    __bf16* Wb1 = (__bf16*)walloc((size_t)256 * 256 * 2);
    __bf16* Wb2 = (__bf16*)walloc((size_t)256 * 128 * 2);
    float* h1   = (float*)walloc((size_t)N * 256 * 4);   // layer1 lin out; reused as layer2 lin out
    float* agg1 = (float*)walloc((size_t)N * 256 * 4);   // layer1 agg; post-act in place
    float* agg2 = (float*)walloc((size_t)N * 128 * 4);
    float* a_s  = (float*)walloc((size_t)N * 2 * 4);
    float* a_d  = (float*)walloc((size_t)N * 2 * 4);
    float* mbuf = (float*)walloc((size_t)N * 2 * 4);
    float* sbuf = (float*)walloc((size_t)N * 2 * 4);
    float* ex   = (float*)walloc((size_t)Etot * 2 * 4);

    const int T = 256;
    auto cdiv = [](int a, int b) { return (a + b - 1) / b; };

    k_cvt_bf16<<<cdiv(256 * 256, T), T, 0, stream>>>(W1, Wb1, 256 * 256);
    k_cvt_bf16<<<cdiv(256 * 128, T), T, 0, stream>>>(W2, Wb2, 256 * 128);

    // ---------------- layer 1 ----------------
    k_fill<<<cdiv(N * 2, T), T, 0, stream>>>(a_s, 0.f, N * 2);
    k_fill<<<cdiv(N * 2, T), T, 0, stream>>>(a_d, 0.f, N * 2);
    {
        dim3 g(cdiv(N, 128), 4);
        k_gemm_att<<<g, 256, 0, stream>>>(x, Wb1, N, 256, 128, as1, ad1, h1, a_s, a_d);
    }
    k_fill<<<cdiv(N * 2, T), T, 0, stream>>>(mbuf, -INFINITY, N * 2);
    k_fill<<<cdiv(N * 2, T), T, 0, stream>>>(sbuf, 0.f, N * 2);
    k_logit_max<<<cdiv(Etot, T), T, 0, stream>>>(ei, E, N, a_s, a_d, mbuf);
    k_expsum<<<cdiv(Etot, T), T, 0, stream>>>(ei, E, N, a_s, a_d, mbuf, ex, sbuf);
    k_fill<<<cdiv(N * 256, T), T, 0, stream>>>(agg1, 0.f, N * 256);
    k_aggregate<<<cdiv(Etot, 8), 256, 0, stream>>>(ei, E, N, 256, 128, h1, ex, sbuf, agg1);
    k_post1<<<cdiv(N * 256, T), T, 0, stream>>>(agg1, b1, bng, bnb, bnm, bnv, N * 256);

    // ---------------- layer 2 ----------------
    k_fill<<<cdiv(N * 2, T), T, 0, stream>>>(a_s, 0.f, N * 2);
    k_fill<<<cdiv(N * 2, T), T, 0, stream>>>(a_d, 0.f, N * 2);
    {
        dim3 g(cdiv(N, 128), 2);
        k_gemm_att<<<g, 256, 0, stream>>>(agg1, Wb2, N, 128, 64, as2, ad2, h1, a_s, a_d);
    }
    k_fill<<<cdiv(N * 2, T), T, 0, stream>>>(mbuf, -INFINITY, N * 2);
    k_fill<<<cdiv(N * 2, T), T, 0, stream>>>(sbuf, 0.f, N * 2);
    k_logit_max<<<cdiv(Etot, T), T, 0, stream>>>(ei, E, N, a_s, a_d, mbuf);
    k_expsum<<<cdiv(Etot, T), T, 0, stream>>>(ei, E, N, a_s, a_d, mbuf, ex, sbuf);
    k_fill<<<cdiv(N * 128, T), T, 0, stream>>>(agg2, 0.f, N * 128);
    k_aggregate<<<cdiv(Etot, 8), 256, 0, stream>>>(ei, E, N, 128, 64, h1, ex, sbuf, agg2);
    k_final<<<cdiv(N, 8), 256, 0, stream>>>(agg2, b2, out, N);
}
